// WaveletScatteringTransform_70007966925448
// MI455X (gfx1250) — matrix operand order repaired and checked by reading, not compiled, not run on hardware
//
#include <hip/hip_runtime.h>
#include <math.h>

// ---------------- problem constants ----------------
constexpr int TT    = 16000;    // signal length = 125 * 128
constexpr int NN1   = 125;      // stage-3 DFT length (padded to 128)
constexpr int BATCH = 64;
constexpr int NPSI  = 36;
constexpr int NCH   = 37;       // S0 + 36 S1 channels
constexpr int TOUT  = 250;      // 16000 / 64
constexpr int LDB   = 136;      // LDS row stride (bf16 elems) for 128x128 data planes
constexpr int LDS_S = 40;       // LDS row stride (bf16 elems) for 128x16 strip planes

// ---------------- workspace layout (float offsets) ----------------
// 12 bf16 table planes (hi/lo pairs), each 128*128 bf16 = 8192 floats:
//  0:F128R_h 1:F128R_l 2:F128I_h 3:F128I_l 4:F128IN_h 5:F128IN_l
//  6:F125R_h 7:F125R_l 8:F125I_h 9:F125I_l 10:F125IN_h 11:F125IN_l
constexpr int TBPL     = 128*128/2;               // floats per bf16 plane
constexpr int OF_TAB   = 0;
constexpr int OF_TWR   = OF_TAB + 12*TBPL;        // f32 twiddles 125x128
constexpr int OF_TWI   = OF_TWR + 125*128;
constexpr int OF_CS250 = OF_TWI + 125*128;        // cos/sin table, 250 x 2
constexpr int OF_XHAT  = OF_CS250 + 512;          // [B][T][2] f32 interleaved
constexpr int OF_SBUF  = OF_XHAT + BATCH*TT*2;    // [B][37][250] f32
constexpr int WS_FLOATS = OF_SBUF + BATCH*NCH*TOUT;   // ~10.6 MB

// ---------------- WMMA types ----------------
typedef __attribute__((ext_vector_type(16))) __bf16 v16bf;
typedef __attribute__((ext_vector_type(8)))  __bf16 v8bf;
typedef __attribute__((ext_vector_type(8)))  float  v8f;

struct FragBF { v16bf hi, lo; };

// two aligned 16B loads -> one v16bf operand half
__device__ inline v16bf load16(const __bf16* p0, const __bf16* p1) {
  v8bf a = *reinterpret_cast<const v8bf*>(p0);
  v8bf b = *reinterpret_cast<const v8bf*>(p1);
  return __builtin_shufflevector(a, b, 0,1,2,3,4,5,6,7,8,9,10,11,12,13,14,15);
}

// A fragment (16x32). ISA layout: lane&15 = row; lane>=16 -> K+8;
// vector elems 0..7 <-> K = kb+0..7, elems 8..15 <-> K = kb+16..23 (contiguous!)
__device__ inline FragBF frag_a(const __bf16* hi, const __bf16* lo, int ld, int m0, int k0) {
  int lane = threadIdx.x & 31;
  int off = (m0 + (lane & 15)) * ld + k0 + ((lane & 16) ? 8 : 0);
  FragBF f;
  f.hi = load16(hi + off, hi + off + 16);
  f.lo = load16(lo + off, lo + off + 16);
  return f;
}

// B fragment (32x16). ISA layout: lane&15 = col N; lane>=16 -> K+8;
// elems 0..7 <-> K = k0+0..7, elems 8..15 <-> K = k0+16..23 (same as A)
__device__ inline FragBF frag_b(const __bf16* hi, const __bf16* lo, int ld, int k0, int n0) {
  int lane = threadIdx.x & 31;
  int n = n0 + (lane & 15);
  int kb = k0 + ((lane & 16) ? 8 : 0);
  FragBF f;
#pragma unroll
  for (int s = 0; s < 16; ++s) {
    int off = (kb + (s < 8 ? s : s + 8)) * ld + n;
    f.hi[s] = hi[off];
    f.lo[s] = lo[off];
  }
  return f;
}

// hi/lo bf16 3-product split: ~fp32 accuracy on the bf16 WMMA pipe
__device__ inline v8f mac3(const FragBF& a, const FragBF& b, v8f c) {
  c = __builtin_amdgcn_wmma_f32_16x16x32_bf16(false, a.hi, false, b.hi, (short)0, c, false, false);
  c = __builtin_amdgcn_wmma_f32_16x16x32_bf16(false, a.hi, false, b.lo, (short)0, c, false, false);
  c = __builtin_amdgcn_wmma_f32_16x16x32_bf16(false, a.lo, false, b.hi, (short)0, c, false, false);
  return c;
}

// producer-side hi/lo split (done once per element)
__device__ inline void store_hilo(__bf16* hi, __bf16* lo, int idx, float x) {
  __bf16 h = (__bf16)x;
  hi[idx] = h;
  lo[idx] = (__bf16)(x - (float)h);
}

// Stage 1: Y[j1][k2] = sum_j2 data[j1][j2] * F128[j2][k2].
// A data from LDS bf16 hi/lo planes (aIh==nullptr -> real input).
// B table planes (br, bi, bn) with bn = -bi realizing Cre -= Ai*Bi.
__device__ inline void stage1_cplx(const __bf16* aRh, const __bf16* aRl,
                                   const __bf16* aIh, const __bf16* aIl,
                                   const __bf16* brh, const __bf16* brl,
                                   const __bf16* bih, const __bf16* bil,
                                   const __bf16* bnh, const __bf16* bnl,
                                   int n0, v8f cre[2], v8f cim[2]) {
  int w = (threadIdx.x >> 5) & 3;
  v8f z = {};
  cre[0] = z; cre[1] = z; cim[0] = z; cim[1] = z;
  for (int k0 = 0; k0 < 128; k0 += 32) {
    FragBF br = frag_b(brh, brl, 128, k0, n0);
    FragBF bi = frag_b(bih, bil, 128, k0, n0);
    FragBF bn = frag_b(bnh, bnl, 128, k0, n0);
#pragma unroll
    for (int t = 0; t < 2; ++t) {
      int m0 = 16 * (w + 4 * t);
      FragBF ar = frag_a(aRh, aRl, LDB, m0, k0);
      cre[t] = mac3(ar, br, cre[t]);
      cim[t] = mac3(ar, bi, cim[t]);
      if (aIh) {
        FragBF ai = frag_a(aIh, aIl, LDB, m0, k0);
        cre[t] = mac3(ai, bn, cre[t]);
        cim[t] = mac3(ai, br, cim[t]);
      }
    }
  }
}

// Stage 3: Z[k1][s] = sum_j1 F125[k1][j1] * Ystrip[j1][s]  (F125 symmetric)
__device__ inline void stage3_cplx(const __bf16* arh, const __bf16* arl,
                                   const __bf16* aih, const __bf16* ail,
                                   const __bf16* anh, const __bf16* anl,
                                   const __bf16* srh, const __bf16* srl,
                                   const __bf16* sih, const __bf16* sil,
                                   v8f cre[2], v8f cim[2]) {
  int w = (threadIdx.x >> 5) & 3;
  v8f z = {};
  cre[0] = z; cre[1] = z; cim[0] = z; cim[1] = z;
  for (int k0 = 0; k0 < 128; k0 += 32) {
    FragBF br = frag_b(srh, srl, LDS_S, k0, 0);
    FragBF bi = frag_b(sih, sil, LDS_S, k0, 0);
#pragma unroll
    for (int t = 0; t < 2; ++t) {
      int m0 = 16 * (w + 4 * t);
      FragBF ar = frag_a(arh, arl, 128, m0, k0);
      FragBF ai = frag_a(aih, ail, 128, m0, k0);
      FragBF an = frag_a(anh, anl, 128, m0, k0);
      cre[t] = mac3(ar, br, cre[t]);
      cre[t] = mac3(an, bi, cre[t]);   // Cre -= Ai*Bi via negated table
      cim[t] = mac3(ar, bi, cim[t]);
      cim[t] = mac3(ai, br, cim[t]);
    }
  }
}

// Apply twiddle to stage-1 accumulators, store bf16 hi/lo strip to LDS
__device__ inline void twiddle_store(v8f cre[2], v8f cim[2],
                                     const float* twr, const float* twi, float twsign,
                                     int n0, __bf16* ysRh, __bf16* ysRl,
                                     __bf16* ysIh, __bf16* ysIl) {
  int w = (threadIdx.x >> 5) & 3;
  int lane = threadIdx.x & 31;
  int colL = lane & 15;
  int col = n0 + colL;
  int rowB = (lane & 16) ? 8 : 0;
#pragma unroll
  for (int t = 0; t < 2; ++t) {
    int m0 = 16 * (w + 4 * t);
#pragma unroll
    for (int v = 0; v < 8; ++v) {
      int row = m0 + rowB + v;          // j1
      float yr = cre[t][v], yi = cim[t][v];
      float tr = 1.f, ti = 0.f;
      if (row < NN1) { tr = twr[row * 128 + col]; ti = twsign * twi[row * 128 + col]; }
      store_hilo(ysRh, ysRl, row * LDS_S + colL, yr * tr - yi * ti);
      store_hilo(ysIh, ysIl, row * LDS_S + colL, yr * ti + yi * tr);
    }
  }
}

__device__ inline float freq_of(int k) {
  float f = (float)k * (1.0f / (float)TT);
  return (2 * k < TT) ? f : f - 1.0f;
}

// 250-point inverse DFT (real part) via LDS cos/sin table; 1/T final scale
__device__ inline void idft250_store(const float* sf, const float* cs, float* dst) {
  for (int m = threadIdx.x; m < TOUT; m += blockDim.x) {
    float acc = 0.f;
    for (int q = 0; q < TOUT; ++q) {
      int r = (q * m) % TOUT;
      acc += sf[2 * q] * cs[2 * r] - sf[2 * q + 1] * cs[2 * r + 1];
    }
    dst[m] = acc * (1.0f / (float)TT);
  }
}

// ---------------- K0: tables (f64 trig, bf16 hi/lo planes) ----------------
__global__ void k_setup(float* __restrict__ ws) {
  int idx = blockIdx.x * blockDim.x + threadIdx.x;
  __bf16* TB = (__bf16*)(ws + OF_TAB);
  auto put = [&](int plane, int i, float v) {
    __bf16 h = (__bf16)v;
    TB[plane * 16384 + i] = h;
    TB[(plane + 1) * 16384 + i] = (__bf16)(v - (float)h);
  };
  const double PI = 3.14159265358979323846;
  if (idx < 128 * 128) {
    int j = idx >> 7, k = idx & 127;
    {
      int m = (j * k) & 127;
      double a = -2.0 * PI * (double)m / 128.0;
      float cr = (float)cos(a), si = (float)sin(a);
      put(0, idx, cr); put(2, idx, si); put(4, idx, -si);
    }
    if (j < NN1 && k < NN1) {
      int m = (j * k) % NN1;
      double a = -2.0 * PI * (double)m / (double)NN1;
      float cr = (float)cos(a), si = (float)sin(a);
      put(6, idx, cr); put(8, idx, si); put(10, idx, -si);
    } else {
      put(6, idx, 0.f); put(8, idx, 0.f); put(10, idx, 0.f);
    }
  }
  if (idx < NN1 * 128) {
    int j1 = idx >> 7, k2 = idx & 127;
    int m = (int)(((long long)j1 * k2) % TT);
    double a = -2.0 * PI * (double)m / (double)TT;
    ws[OF_TWR + idx] = (float)cos(a);
    ws[OF_TWI + idx] = (float)sin(a);
  }
  if (idx < TOUT) {
    double a = 2.0 * PI * (double)idx / (double)TOUT;
    ws[OF_CS250 + 2 * idx]     = (float)cos(a);
    ws[OF_CS250 + 2 * idx + 1] = (float)sin(a);
  }
}

// table plane accessor
#define TPL(i) (TB + (i) * 16384)

// ---------------- K1: forward FFT of x -> XHAT ----------------
__global__ __launch_bounds__(128) void k_fft_x(const float* __restrict__ x, float* __restrict__ ws) {
  const __bf16* TB = (const __bf16*)(ws + OF_TAB);
  const float* TWR = ws + OF_TWR;
  const float* TWI = ws + OF_TWI;
  float* XHAT = ws + OF_XHAT;

  __shared__ __bf16 P0h[128 * LDB];
  __shared__ __bf16 P0l[128 * LDB];
  __shared__ __bf16 YsRh[128 * LDS_S];
  __shared__ __bf16 YsRl[128 * LDS_S];
  __shared__ __bf16 YsIh[128 * LDS_S];
  __shared__ __bf16 YsIl[128 * LDS_S];

  const int b = blockIdx.x;
  const int tid = threadIdx.x;
  const int lane = tid & 31;
  const int w = tid >> 5;

  // x_mat[j1][j2] = x[b][j1 + 125*j2]; pad rows 125..127 = 0
  for (int idx = tid; idx < TT; idx += 128) {
    int j1 = idx % NN1, j2 = idx / NN1;
    store_hilo(P0h, P0l, j1 * LDB + j2, x[b * TT + idx]);
  }
  for (int idx = tid; idx < 3 * 128; idx += 128) {
    int o = (NN1 + idx / 128) * LDB + (idx % 128);
    P0h[o] = (__bf16)0.f; P0l[o] = (__bf16)0.f;
  }
  __syncthreads();

  v8f cre[2], cim[2];
  for (int s = 0; s < 8; ++s) {
    int n0 = 16 * s;
    stage1_cplx(P0h, P0l, nullptr, nullptr,
                TPL(0), TPL(1), TPL(2), TPL(3), TPL(4), TPL(5),
                n0, cre, cim);
    twiddle_store(cre, cim, TWR, TWI, 1.0f, n0, YsRh, YsRl, YsIh, YsIl);
    __syncthreads();
    stage3_cplx(TPL(6), TPL(7), TPL(8), TPL(9), TPL(10), TPL(11),
                YsRh, YsRl, YsIh, YsIl, cre, cim);
    // store: X[k2 + 128*k1]
    int colL = lane & 15, rowB = (lane & 16) ? 8 : 0;
#pragma unroll
    for (int t = 0; t < 2; ++t) {
      int m0 = 16 * (w + 4 * t);
#pragma unroll
      for (int v = 0; v < 8; ++v) {
        int row = m0 + rowB + v;
        if (row < NN1) {
          int k = (n0 + colL) + 128 * row;
          XHAT[(b * TT + k) * 2]     = cre[t][v];
          XHAT[(b * TT + k) * 2 + 1] = cim[t][v];
        }
      }
    }
    __syncthreads();
  }
}

// ---------------- K2: per (batch, channel) scattering ----------------
__global__ __launch_bounds__(128) void k_scatter(float* __restrict__ ws) {
  const __bf16* TB = (const __bf16*)(ws + OF_TAB);
  const float* TWR = ws + OF_TWR;
  const float* TWI = ws + OF_TWI;
  const float* XHAT = ws + OF_XHAT;
  float* SBUF = ws + OF_SBUF;

  __shared__ __bf16 P0h[128 * LDB];   // H real
  __shared__ __bf16 P0l[128 * LDB];
  __shared__ __bf16 P1h[128 * LDB];   // H imag
  __shared__ __bf16 P1l[128 * LDB];
  __shared__ __bf16 P2h[128 * LDB];   // U = |u|
  __shared__ __bf16 P2l[128 * LDB];
  __shared__ __bf16 YsRh[128 * LDS_S];
  __shared__ __bf16 YsRl[128 * LDS_S];
  __shared__ __bf16 YsIh[128 * LDS_S];
  __shared__ __bf16 YsIl[128 * LDS_S];
  __shared__ float sfold[2 * TOUT];
  __shared__ float cs[2 * TOUT];

  const int b = blockIdx.x;
  const int ci = blockIdx.y;
  const int tid = threadIdx.x;
  const int lane = tid & 31;
  const int w = tid >> 5;

  const float sphi = 0.35f / 64.f;
  const float inv2sphi = 1.f / (2.f * sphi * sphi);

  for (int q = tid; q < 2 * TOUT; q += 128) {
    sfold[q] = 0.f;
    cs[q] = ws[OF_CS250 + q];
  }

  if (ci == NPSI) {            // ---- S0 channel: fold XHAT*phi directly ----
    __syncthreads();
    for (int k = tid; k < TT; k += 128) {
      float f = freq_of(k);
      float g = __expf(-f * f * inv2sphi);
      float xr = XHAT[(b * TT + k) * 2];
      float xi = XHAT[(b * TT + k) * 2 + 1];
      int q = k % TOUT;
      atomicAdd(&sfold[2 * q], xr * g);
      atomicAdd(&sfold[2 * q + 1], xi * g);
    }
    __syncthreads();
    idft250_store(sfold, cs, SBUF + (b * NCH + 0) * TOUT);
    return;
  }

  // ---- S1 channel ci ----
  const float xi_c = 0.35f * exp2f(-(float)ci / 6.0f);
  const float rq = exp2f(1.0f / 6.0f);
  const float sig = xi_c * (rq - 1.f) / (rq + 1.f);
  const float inv2s = 1.f / (2.f * sig * sig);

  // build H = XHAT * psi_hat / T as bf16 hi/lo planes
  for (int k = tid; k < TT; k += 128) {
    __builtin_prefetch(&XHAT[(b * TT + ((k + 512 < TT) ? k + 512 : k)) * 2], 0, 1);
    float f = freq_of(k);
    float d = f - xi_c;
    float g = __expf(-d * d * inv2s) * (1.0f / (float)TT);
    float xr = XHAT[(b * TT + k) * 2];
    float xim = XHAT[(b * TT + k) * 2 + 1];
    int o = (k % NN1) * LDB + (k / NN1);
    store_hilo(P0h, P0l, o, xr * g);
    store_hilo(P1h, P1l, o, xim * g);
  }
  for (int idx = tid; idx < 3 * 128; idx += 128) {
    int o = (NN1 + idx / 128) * LDB + (idx % 128);
    P0h[o] = (__bf16)0.f; P0l[o] = (__bf16)0.f;
    P1h[o] = (__bf16)0.f; P1l[o] = (__bf16)0.f;
    P2h[o] = (__bf16)0.f; P2l[o] = (__bf16)0.f;
  }
  __syncthreads();

  v8f cre[2], cim[2];
  const int colL = lane & 15, rowB = (lane & 16) ? 8 : 0;

  // ---- inverse FFT of H (conj tables: i <-> -i plane swap), U=|u| into P2 ----
  for (int s = 0; s < 8; ++s) {
    int n0 = 16 * s;
    stage1_cplx(P0h, P0l, P1h, P1l,
                TPL(0), TPL(1), TPL(4), TPL(5), TPL(2), TPL(3),   // swapped i/in
                n0, cre, cim);
    twiddle_store(cre, cim, TWR, TWI, -1.0f, n0, YsRh, YsRl, YsIh, YsIl);
    __syncthreads();
    stage3_cplx(TPL(6), TPL(7), TPL(10), TPL(11), TPL(8), TPL(9), // swapped i/in
                YsRh, YsRl, YsIh, YsIl, cre, cim);
#pragma unroll
    for (int t = 0; t < 2; ++t) {
      int m0 = 16 * (w + 4 * t);
#pragma unroll
      for (int v = 0; v < 8; ++v) {
        int row = m0 + rowB + v;
        if (row < NN1) {
          int tt = (n0 + colL) + 128 * row;
          float zr = cre[t][v], zi = cim[t][v];
          store_hilo(P2h, P2l, (tt % NN1) * LDB + (tt / NN1), sqrtf(zr * zr + zi * zi));
        }
      }
    }
    __syncthreads();
  }

  // ---- forward FFT of U, * phi_hat, fold mod 250 via ds_add_f32 ----
  for (int s = 0; s < 8; ++s) {
    int n0 = 16 * s;
    stage1_cplx(P2h, P2l, nullptr, nullptr,
                TPL(0), TPL(1), TPL(2), TPL(3), TPL(4), TPL(5),
                n0, cre, cim);
    twiddle_store(cre, cim, TWR, TWI, 1.0f, n0, YsRh, YsRl, YsIh, YsIl);
    __syncthreads();
    stage3_cplx(TPL(6), TPL(7), TPL(8), TPL(9), TPL(10), TPL(11),
                YsRh, YsRl, YsIh, YsIl, cre, cim);
#pragma unroll
    for (int t = 0; t < 2; ++t) {
      int m0 = 16 * (w + 4 * t);
#pragma unroll
      for (int v = 0; v < 8; ++v) {
        int row = m0 + rowB + v;
        if (row < NN1) {
          int k = (n0 + colL) + 128 * row;
          float f = freq_of(k);
          float g = __expf(-f * f * inv2sphi);
          int q = k % TOUT;
          atomicAdd(&sfold[2 * q], cre[t][v] * g);
          atomicAdd(&sfold[2 * q + 1], cim[t][v] * g);
        }
      }
    }
    __syncthreads();
  }

  idft250_store(sfold, cs, SBUF + (b * NCH + ci + 1) * TOUT);
}

// ---------------- K3: 3-band channel means -> output (64,3,250,1) ----------------
__global__ void k_bands(const float* __restrict__ ws, float* __restrict__ out) {
  int o = blockIdx.x * blockDim.x + threadIdx.x;
  if (o >= BATCH * 3 * TOUT) return;
  const float* SBUF = ws + OF_SBUF;
  int m = o % TOUT;
  int band = (o / TOUT) % 3;
  int b = o / (3 * TOUT);
  int c0 = (band == 0) ? 0 : (band == 1) ? 12 : 24;
  int c1 = (band == 0) ? 12 : (band == 1) ? 24 : 37;
  float s = 0.f;
  for (int c = c0; c < c1; ++c) s += SBUF[(b * NCH + c) * TOUT + m];
  out[o] = s / (float)(c1 - c0);
}

// ---------------- host entry ----------------
extern "C" void kernel_launch(void* const* d_in, const int* in_sizes, int n_in,
                              void* d_out, int out_size, void* d_ws, size_t ws_size,
                              hipStream_t stream) {
  const float* x = (const float*)d_in[0];     // (64,1,16000) f32, flat
  float* out = (float*)d_out;                 // (64,3,250,1) f32 = 48000
  float* ws = (float*)d_ws;                   // needs ~10.6 MB (WS_FLOATS*4 bytes)
  (void)in_sizes; (void)n_in; (void)out_size; (void)ws_size;

  k_setup<<<(128 * 128 + 255) / 256, 256, 0, stream>>>(ws);
  k_fft_x<<<BATCH, 128, 0, stream>>>(x, ws);
  k_scatter<<<dim3(BATCH, NCH), 128, 0, stream>>>(ws);
  k_bands<<<(BATCH * 3 * TOUT + 255) / 256, 256, 0, stream>>>(ws, out);
}
